// Rnnt_28862180229193
// MI455X (gfx1250) — compile-verified, matrix operand order checked by reading, not run on hardware
//
#include <hip/hip_runtime.h>
#include <utility>

// ---------------------------------------------------------------------------
// CDNA5 (gfx1250) RNN-T: all GEMMs via v_wmma_f32_16x16x32_bf16 (wave32).
// ---------------------------------------------------------------------------

typedef __attribute__((ext_vector_type(16))) __bf16 v16bf;
typedef __attribute__((ext_vector_type(8)))  float  v8f;

union FragBF { v16bf v; unsigned short u[16]; uint4 q[2]; };
union AccF   { v8f   v; float f[8]; };

__device__ __forceinline__ unsigned short f32_to_bf16(float x) {
  unsigned int b = __float_as_uint(x);
  b += 0x7FFFu + ((b >> 16) & 1u);           // round-to-nearest-even
  return (unsigned short)(b >> 16);
}
__device__ __forceinline__ float sigm(float x) { return 1.0f / (1.0f + __expf(-x)); }

// --------------------------- utility kernels -------------------------------

__global__ void fill_u16_kernel(unsigned short* p, long n, unsigned short v) {
  long i = (long)blockIdx.x * blockDim.x + threadIdx.x;
  long s = (long)gridDim.x * blockDim.x;
  for (; i < n; i += s) p[i] = v;
}
__global__ void fill_f32_kernel(float* p, long n, float v) {
  long i = (long)blockIdx.x * blockDim.x + threadIdx.x;
  long s = (long)gridDim.x * blockDim.x;
  for (; i < n; i += s) p[i] = v;
}

// f32 -> bf16 with row pitch change and zero column padding
__global__ void cvt_pad_bf16_kernel(const float* __restrict__ src, long spitch,
                                    unsigned short* __restrict__ dst, long dpitch,
                                    long rows, long cols) {
  long n = rows * dpitch;
  long i = (long)blockIdx.x * blockDim.x + threadIdx.x;
  long s = (long)gridDim.x * blockDim.x;
  for (; i < n; i += s) {
    long r = i / dpitch, c = i - r * dpitch;
    float v = (c < cols) ? src[r * spitch + c] : 0.0f;
    dst[i] = f32_to_bf16(v);
  }
}

// stack_time: out rows (t2*4+b) x 2048 ; out[.., f*1024+h] = hs[((t2*2+f)*4+b)*1024+h]
__global__ void stack_time_kernel(const float* __restrict__ hs, float* __restrict__ out) {
  const long n = 1024l * 2048l;
  long i = (long)blockIdx.x * blockDim.x + threadIdx.x;
  long s = (long)gridDim.x * blockDim.x;
  for (; i < n; i += s) {
    long row = i >> 11, col = i & 2047;
    long t2 = row >> 2, b = row & 3;
    long f = col >> 10, h = col & 1023;
    out[i] = hs[(((t2 * 2 + f) * 4) + b) * 1024 + h];
  }
}

// embed gather: out rows (u*4+b) x 320, y is (B=4,U=64)
__global__ void embed_gather_kernel(const int* __restrict__ y,
                                    const float* __restrict__ emb,
                                    float* __restrict__ out) {
  const long n = 256l * 320l;
  long i = (long)blockIdx.x * blockDim.x + threadIdx.x;
  long s = (long)gridDim.x * blockDim.x;
  for (; i < n; i += s) {
    long row = i / 320, h = i - row * 320;
    int u = (int)(row >> 2), b = (int)(row & 3);
    int id = y[b * 64 + u];
    out[i] = emb[(long)id * 320 + h];
  }
}

// --------------------------- WMMA GEMM -------------------------------------
// C[M x N] = A[M x K] * W[N x K]^T (+ bias0[n] + bias1[n])
// grid.x = N/64 (4 n-tiles per wave), grid.y = M/16, block = 32 (one wave)
__global__ __launch_bounds__(32)
void gemm_wmma_kernel(const unsigned short* __restrict__ A, long lda,
                      const unsigned short* __restrict__ W, long ldw,
                      float* __restrict__ C, long ldc, int K,
                      const float* __restrict__ bias0,
                      const float* __restrict__ bias1) {
  const int lane = threadIdx.x;
  const int m0 = blockIdx.y * 16;
  const int n0 = blockIdx.x * 64;
  const int rc = lane & 15;          // A row / B col within tile
  const int half = lane >> 4;

  AccF acc[4];
#pragma unroll
  for (int t = 0; t < 4; ++t)
#pragma unroll
    for (int r = 0; r < 8; ++r) acc[t].f[r] = 0.0f;

  const unsigned short* Arow = A + (size_t)(m0 + rc) * lda;
  const unsigned short* Wrow[4];
#pragma unroll
  for (int t = 0; t < 4; ++t) Wrow[t] = W + (size_t)(n0 + t * 16 + rc) * ldw;

  for (int k0 = 0; k0 < K; k0 += 32) {
    const int kA = k0 + half * 8, kB = kA + 16;
    FragBF a;
    a.q[0] = *(const uint4*)(Arow + kA);
    a.q[1] = *(const uint4*)(Arow + kB);
#pragma unroll
    for (int t = 0; t < 4; ++t) {
      FragBF b;
      b.q[0] = *(const uint4*)(Wrow[t] + kA);
      b.q[1] = *(const uint4*)(Wrow[t] + kB);
      acc[t].v = __builtin_amdgcn_wmma_f32_16x16x32_bf16(
          false, a.v, false, b.v, (short)0, acc[t].v, false, false);
    }
  }

  const int mbase = m0 + half * 8;
#pragma unroll
  for (int t = 0; t < 4; ++t) {
    const int n = n0 + t * 16 + rc;
    float bs = 0.0f;
    if (bias0) bs += bias0[n];
    if (bias1) bs += bias1[n];
#pragma unroll
    for (int r = 0; r < 8; ++r)
      C[(size_t)(mbase + r) * ldc + n] = acc[t].f[r] + bs;
  }
}

// --------------------------- LSTM recurrent step ---------------------------
// One wave per 16 hidden units; computes all 4 gates (cols n, H+n, 2H+n, 3H+n)
// of g = xg[t] + h_prev @ Whh^T, applies nonlinearity, writes h (f32 + bf16), c.
// h_in/h_out: 16 x H bf16 (rows 4..15 stay zero). grid.x = H/16, block = 32.
__global__ __launch_bounds__(32)
void lstm_step_kernel(const unsigned short* __restrict__ h_in,
                      const unsigned short* __restrict__ Whh,   // 4H x H bf16
                      const float* __restrict__ xg_t,           // 4 x 4H (row = batch)
                      float* __restrict__ c,                    // 4 x H
                      float* __restrict__ h_out_f,              // 4 x H (hs slice)
                      unsigned short* __restrict__ h_out_bf,    // 16 x H
                      int H) {
  const int lane = threadIdx.x;
  const int n0 = blockIdx.x * 16;
  const int col = lane & 15;
  const int half = lane >> 4;
  const int n = n0 + col;
  const long H4 = 4l * H;

  AccF acc[4];
  const int mbase = half * 8;
#pragma unroll
  for (int g = 0; g < 4; ++g)
#pragma unroll
    for (int r = 0; r < 8; ++r) {
      const int m = mbase + r;
      acc[g].f[r] = (m < 4) ? xg_t[(long)m * H4 + (long)g * H + n] : 0.0f;
    }

  const unsigned short* Arow = h_in + (size_t)col * H;
  const unsigned short* Wrow[4];
#pragma unroll
  for (int g = 0; g < 4; ++g) Wrow[g] = Whh + ((size_t)g * H + n) * H;

  for (int k0 = 0; k0 < H; k0 += 32) {
    const int kA = k0 + half * 8, kB = kA + 16;
    FragBF a;
    a.q[0] = *(const uint4*)(Arow + kA);
    a.q[1] = *(const uint4*)(Arow + kB);
#pragma unroll
    for (int g = 0; g < 4; ++g) {
      FragBF b;
      b.q[0] = *(const uint4*)(Wrow[g] + kA);
      b.q[1] = *(const uint4*)(Wrow[g] + kB);
      acc[g].v = __builtin_amdgcn_wmma_f32_16x16x32_bf16(
          false, a.v, false, b.v, (short)0, acc[g].v, false, false);
    }
  }

  if (lane < 16) {
#pragma unroll
    for (int r = 0; r < 4; ++r) {          // batch rows 0..3 live in VGPRs 0..3, lanes 0..15
      const float iv = sigm(acc[0].f[r]);
      const float fv = sigm(acc[1].f[r]);
      const float gv = tanhf(acc[2].f[r]);
      const float ov = sigm(acc[3].f[r]);
      const size_t ci = (size_t)r * H + n;
      const float cn = fv * c[ci] + iv * gv;
      c[ci] = cn;
      const float hv = ov * tanhf(cn);
      h_out_f[ci] = hv;
      h_out_bf[ci] = f32_to_bf16(hv);
    }
  }
}

// --------------------------- fused joint -----------------------------------
// logits[b,t,u,k] = relu(zf[t*4+b] + zg[u*4+b] + jb1) . jw2[k] + jb2[k]
// A (M=65536 x K=512) built on the fly in bf16; B = jw2 padded to 32x512.
// grid.x = 65536/16 = 4096, block = 32.
__device__ __forceinline__ void relu_chunk_bf16(const float* zf, const float* zg,
                                                const float* b1, int k,
                                                unsigned short* dst) {
#pragma unroll
  for (int i = 0; i < 8; ++i) {
    const float v = zf[k + i] + zg[k + i] + b1[k + i];
    dst[i] = f32_to_bf16(v > 0.0f ? v : 0.0f);
  }
}

__global__ __launch_bounds__(32)
void joint_wmma_kernel(const float* __restrict__ zf,   // 1024 x 512 (rows t*4+b)
                       const float* __restrict__ zg,   // 256 x 512  (rows u*4+b)
                       const float* __restrict__ jb1,  // 512
                       const unsigned short* __restrict__ w2,  // 32 x 512 bf16 (padded)
                       const float* __restrict__ jb2,  // 29
                       float* __restrict__ out) {      // (4,256,64,29)
  const int lane = threadIdx.x;
  const int m0 = blockIdx.x * 16;
  const int col = lane & 15;
  const int half = lane >> 4;

  const int m = m0 + col;
  const int u = m & 63, t = (m >> 6) & 255, b = m >> 14;
  const float* zfrow = zf + (size_t)(t * 4 + b) * 512;
  const float* zgrow = zg + (size_t)(u * 4 + b) * 512;

  AccF acc[2];
#pragma unroll
  for (int nt = 0; nt < 2; ++nt)
#pragma unroll
    for (int r = 0; r < 8; ++r) acc[nt].f[r] = 0.0f;

  const unsigned short* Wrow[2];
#pragma unroll
  for (int nt = 0; nt < 2; ++nt) Wrow[nt] = w2 + (size_t)(nt * 16 + col) * 512;

  for (int k0 = 0; k0 < 512; k0 += 32) {
    const int kA = k0 + half * 8, kB = kA + 16;
    FragBF a;
    relu_chunk_bf16(zfrow, zgrow, jb1, kA, &a.u[0]);
    relu_chunk_bf16(zfrow, zgrow, jb1, kB, &a.u[8]);
#pragma unroll
    for (int nt = 0; nt < 2; ++nt) {
      FragBF bb;
      bb.q[0] = *(const uint4*)(Wrow[nt] + kA);
      bb.q[1] = *(const uint4*)(Wrow[nt] + kB);
      acc[nt].v = __builtin_amdgcn_wmma_f32_16x16x32_bf16(
          false, a.v, false, bb.v, (short)0, acc[nt].v, false, false);
    }
  }

#pragma unroll
  for (int nt = 0; nt < 2; ++nt) {
    const int n = nt * 16 + col;
    if (n < 29) {
      const float bias = jb2[n];
#pragma unroll
      for (int r = 0; r < 8; ++r) {
        const int mm = m0 + half * 8 + r;
        const int uu = mm & 63, tt = (mm >> 6) & 255, bbq = mm >> 14;
        out[(((size_t)bbq * 256 + tt) * 64 + uu) * 29 + n] = acc[nt].f[r] + bias;
      }
    }
  }
}

// --------------------------- host orchestration ----------------------------

extern "C" void kernel_launch(void* const* d_in, const int* in_sizes, int n_in,
                              void* d_out, int out_size, void* d_ws, size_t ws_size,
                              hipStream_t stream) {
  (void)in_sizes; (void)n_in; (void)out_size; (void)ws_size;

  const float* x_padded = (const float*)d_in[0];
  const int*   y        = (const int*)d_in[2];     // label indices (0..27)
  const float* preW[2][4];
  for (int l = 0; l < 2; ++l) for (int j = 0; j < 4; ++j) preW[l][j]  = (const float*)d_in[3 + l * 4 + j];
  const float* postW[3][4];
  for (int l = 0; l < 3; ++l) for (int j = 0; j < 4; ++j) postW[l][j] = (const float*)d_in[11 + l * 4 + j];
  const float* embed = (const float*)d_in[23];
  const float* predW[2][4];
  for (int l = 0; l < 2; ++l) for (int j = 0; j < 4; ++j) predW[l][j] = (const float*)d_in[24 + l * 4 + j];
  const float* jw1 = (const float*)d_in[32];
  const float* jb1 = (const float*)d_in[33];
  const float* jw2 = (const float*)d_in[34];
  const float* jb2 = (const float*)d_in[35];
  float* out = (float*)d_out;

  // ---- workspace carve (all offsets 256B aligned) ----
  char* wsp = (char*)d_ws;
  auto alloc = [&](size_t bytes) -> char* {
    char* p = wsp; wsp += (bytes + 255) & ~(size_t)255; return p;
  };
  auto aU16 = [&](size_t elems) { return (unsigned short*)alloc(elems * 2); };
  auto aF32 = [&](size_t elems) { return (float*)alloc(elems * 4); };

  // bf16 weights
  unsigned short* preWih1_bf  = aU16(4096ull * 256);
  unsigned short* preWhh1_bf  = aU16(4096ull * 1024);
  unsigned short* preWih2_bf  = aU16(4096ull * 1024);
  unsigned short* preWhh2_bf  = aU16(4096ull * 1024);
  unsigned short* postWih1_bf = aU16(4096ull * 2048);
  unsigned short* postWhh1_bf = aU16(4096ull * 1024);
  unsigned short* postWih2_bf = aU16(4096ull * 1024);
  unsigned short* postWhh2_bf = aU16(4096ull * 1024);
  unsigned short* postWih3_bf = aU16(4096ull * 1024);
  unsigned short* postWhh3_bf = aU16(4096ull * 1024);
  unsigned short* predWih1_bf = aU16(1280ull * 320);
  unsigned short* predWhh1_bf = aU16(1280ull * 320);
  unsigned short* predWih2_bf = aU16(1280ull * 320);
  unsigned short* predWhh2_bf = aU16(1280ull * 320);
  unsigned short* w1f_bf      = aU16(512ull * 1024);
  unsigned short* w1g_bf      = aU16(512ull * 320);
  unsigned short* jw2p_bf     = aU16(32ull * 512);
  // state / activations
  unsigned short* h_bfA = aU16(16ull * 2048);
  unsigned short* h_bfB = aU16(16ull * 2048);
  unsigned short* x_bf  = aU16(2048ull * 2048);
  float* xg       = aF32(2048ull * 4096);
  float* hsA      = aF32(2048ull * 1024);
  float* hsB      = aF32(2048ull * 1024);
  float* stackbuf = aF32(1024ull * 2048);
  float* embbuf   = aF32(256ull * 320);
  float* predh1   = aF32(256ull * 320);
  float* predh2   = aF32(256ull * 320);
  float* cstate   = aF32(4ull * 2048);
  float* zf       = aF32(1024ull * 512);
  float* zg       = aF32(256ull * 512);

  // ---- host launch helpers ----
  auto gsb = [](long n) { unsigned b = (unsigned)((n + 1023) / 1024); return b ? b : 1u; };
  auto lcvt = [&](const float* src, long sp, unsigned short* dst, long dp, long rows, long cols) {
    cvt_pad_bf16_kernel<<<gsb(rows * dp), 256, 0, stream>>>(src, sp, dst, dp, rows, cols);
  };
  auto lfillu = [&](unsigned short* p, long n) {
    fill_u16_kernel<<<gsb(n), 256, 0, stream>>>(p, n, (unsigned short)0);
  };
  auto lfillf = [&](float* p, long n) {
    fill_f32_kernel<<<gsb(n), 256, 0, stream>>>(p, n, 0.0f);
  };
  auto lgemm = [&](const unsigned short* A, long lda, const unsigned short* W, long ldw,
                   float* C, long ldc, int M, int N, int K,
                   const float* b0, const float* b1) {
    gemm_wmma_kernel<<<dim3((unsigned)(N / 64), (unsigned)(M / 16)), 32, 0, stream>>>(
        A, lda, W, ldw, C, ldc, K, b0, b1);
  };
  auto run_steps = [&](int T, int H, const unsigned short* Whh_bf, const float* xg_buf,
                       float* hs_out) {
    lfillu(h_bfA, 16l * H);
    lfillu(h_bfB, 16l * H);
    lfillf(cstate, 4l * H);
    unsigned short* hin = h_bfA;
    unsigned short* hout = h_bfB;
    for (int t = 0; t < T; ++t) {
      lstm_step_kernel<<<(unsigned)(H / 16), 32, 0, stream>>>(
          hin, Whh_bf, xg_buf + (size_t)t * 16 * H, cstate,
          hs_out + (size_t)t * 4 * H, hout, H);
      std::swap(hin, hout);
    }
  };
  // full LSTM layer: in (M=T*4 rows x Ksrc f32) -> hs_out (T x 4 x H f32)
  auto run_layer = [&](const float* in_f32, int M, int Ksrc, int Kpad,
                       const unsigned short* Wih_bf, const unsigned short* Whh_bf,
                       const float* bih, const float* bhh, int T, int H, float* hs_out) {
    lcvt(in_f32, Ksrc, x_bf, Kpad, M, Ksrc);
    lgemm(x_bf, Kpad, Wih_bf, Kpad, xg, 4l * H, M, 4 * H, Kpad, bih, bhh);
    run_steps(T, H, Whh_bf, xg, hs_out);
  };

  // ---- weight conversion (every call; deterministic) ----
  lcvt(preW[0][0], 240,  preWih1_bf, 256,  4096, 240);
  lcvt(preW[0][1], 1024, preWhh1_bf, 1024, 4096, 1024);
  lcvt(preW[1][0], 1024, preWih2_bf, 1024, 4096, 1024);
  lcvt(preW[1][1], 1024, preWhh2_bf, 1024, 4096, 1024);
  lcvt(postW[0][0], 2048, postWih1_bf, 2048, 4096, 2048);
  lcvt(postW[0][1], 1024, postWhh1_bf, 1024, 4096, 1024);
  lcvt(postW[1][0], 1024, postWih2_bf, 1024, 4096, 1024);
  lcvt(postW[1][1], 1024, postWhh2_bf, 1024, 4096, 1024);
  lcvt(postW[2][0], 1024, postWih3_bf, 1024, 4096, 1024);
  lcvt(postW[2][1], 1024, postWhh3_bf, 1024, 4096, 1024);
  lcvt(predW[0][0], 320, predWih1_bf, 320, 1280, 320);
  lcvt(predW[0][1], 320, predWhh1_bf, 320, 1280, 320);
  lcvt(predW[1][0], 320, predWih2_bf, 320, 1280, 320);
  lcvt(predW[1][1], 320, predWhh2_bf, 320, 1280, 320);
  lcvt(jw1,        1344, w1f_bf, 1024, 512, 1024);   // jw1[:, :1024]
  lcvt(jw1 + 1024, 1344, w1g_bf, 320,  512, 320);    // jw1[:, 1024:]
  lfillu(jw2p_bf, 32l * 512);
  lcvt(jw2, 512, jw2p_bf, 512, 29, 512);

  // ---- encoder pre-stack (T=512) ----
  run_layer(x_padded, 2048, 240, 256, preWih1_bf, preWhh1_bf, preW[0][2], preW[0][3], 512, 1024, hsA);
  run_layer(hsA, 2048, 1024, 1024, preWih2_bf, preWhh2_bf, preW[1][2], preW[1][3], 512, 1024, hsB);

  // ---- time stacking (512 -> 256, H -> 2H) ----
  stack_time_kernel<<<gsb(1024l * 2048), 256, 0, stream>>>(hsB, stackbuf);

  // ---- encoder post-stack (T=256) ----
  run_layer(stackbuf, 1024, 2048, 2048, postWih1_bf, postWhh1_bf, postW[0][2], postW[0][3], 256, 1024, hsA);
  run_layer(hsA, 1024, 1024, 1024, postWih2_bf, postWhh2_bf, postW[1][2], postW[1][3], 256, 1024, hsB);
  run_layer(hsB, 1024, 1024, 1024, postWih3_bf, postWhh3_bf, postW[2][2], postW[2][3], 256, 1024, hsA);
  // f = hsA, rows (t*4+b) x 1024

  // ---- prediction network (U=64, H=320) ----
  embed_gather_kernel<<<gsb(256l * 320), 256, 0, stream>>>(y, embed, embbuf);
  run_layer(embbuf, 256, 320, 320, predWih1_bf, predWhh1_bf, predW[0][2], predW[0][3], 64, 320, predh1);
  run_layer(predh1, 256, 320, 320, predWih2_bf, predWhh2_bf, predW[1][2], predW[1][3], 64, 320, predh2);
  // g = predh2, rows (u*4+b) x 320

  // ---- joint projections ----
  lcvt(hsA, 1024, x_bf, 1024, 1024, 1024);
  lgemm(x_bf, 1024, w1f_bf, 1024, zf, 512, 1024, 512, 1024, nullptr, nullptr);
  lcvt(predh2, 320, x_bf, 320, 256, 320);
  lgemm(x_bf, 320, w1g_bf, 320, zg, 512, 256, 512, 320, nullptr, nullptr);

  // ---- fused relu + final projection -> logits (4,256,64,29) ----
  joint_wmma_kernel<<<4096, 32, 0, stream>>>(zf, zg, jb1, jw2p_bf, jb2, out);
}
